// HapticCodec_70016556859798
// MI455X (gfx1250) — matrix operand order repaired and checked by reading, not compiled, not run on hardware
//
#include <hip/hip_runtime.h>
#include <hip/hip_bf16.h>

// ---------------------------------------------------------------------------
// HapticCodec forward for gfx1250 (MI455X).  wave32, WMMA bf16 16x16x32,
// async global->LDS weight staging.
// ---------------------------------------------------------------------------

#define LEAKY_SLOPE 0.01f
#define GN_EPS 1e-5f
#define NB 64           // batch
#define NT 40000        // input length
#define CDIM 64         // code dim
#define LAT 125         // latent length
#define NCB 4
#define CBSZ 256
#define MAXKDPAD 1152   // max IC*k padded to 32 (128*9)
#define AROWPAD 8       // +16B per LDS row: breaks bank conflicts on column reads

typedef __attribute__((ext_vector_type(16))) __bf16 v16bf;
typedef __attribute__((ext_vector_type(8)))  float  v8f;

static inline int ceil_div_i(int a, int b) { return (a + b - 1) / b; }

__device__ __forceinline__ unsigned lds_addr_of(const void* p) {
    // low 32 bits of a generic pointer into LDS == LDS byte address
    return (unsigned)(unsigned long long)p;
}

// ---------------------------------------------------------------------------
// Weight pack: fp32 (OC, Kd) -> bf16 (OC, KdPad) zero-padded in K.
// ---------------------------------------------------------------------------
__global__ void pack_w_bf16_kernel(const float* __restrict__ w, __bf16* __restrict__ wp,
                                   int Kd, int KdPad, int OC) {
    int i = blockIdx.x * blockDim.x + threadIdx.x;
    if (i >= OC * KdPad) return;
    int oc = i / KdPad;
    int kd = i - oc * KdPad;
    float v = (kd < Kd) ? w[oc * Kd + kd] : 0.0f;
    wp[i] = (__bf16)v;
}

// ---------------------------------------------------------------------------
// Branchless implicit-im2col B-fragment gather.  K is the conv kernel width
// (compile-time -> division by constant); (c, kk) maintained incrementally.
// ---------------------------------------------------------------------------
template<int K>
__device__ __forceinline__ v16bf gather_b(const float* __restrict__ X, int bb, int tt,
                                          int IC, int stride, int pad, int Tin,
                                          int kdStart, int Kd, bool nv) {
    v16bf f;
    int c  = kdStart / K;
    int kk = kdStart - c * K;
    const int tb    = tt * stride - pad;
    const int base0 = bb * IC * Tin;
#pragma unroll
    for (int e = 0; e < 16; ++e) {
        int ts = tb + kk;
        bool ok = nv && ((kdStart + e) < Kd) && (ts >= 0) && (ts < Tin);
        int addr = ok ? (base0 + c * Tin + ts) : 0;
        float v = X[addr];                 // unconditional load, select after
        f[e] = (__bf16)(ok ? v : 0.0f);
        ++kk;
        if (kk == K) { kk = 0; ++c; }
    }
    return f;
}

// ---------------------------------------------------------------------------
// Implicit-GEMM conv1d via v_wmma_f32_16x16x32_bf16.
//   A (OC x KdPad bf16 weights) staged block-wide into LDS with
//   global_load_async_to_lds_b32; each wave owns TWO 16x16 C tiles that share
//   one A fragment -> 2 v_wmma per K-step.
//   Y layout: (NB, OC, Tout) fp32, bias fused.
// ---------------------------------------------------------------------------
template<int K>
__global__ __launch_bounds__(128)
void conv1d_wmma_kernel(const __bf16* __restrict__ Wp, const float* __restrict__ bias,
                        const float* __restrict__ X, float* __restrict__ Y,
                        int IC, int pad, int stride,
                        int Tin, int Tout, int OC, int KdPad, int N) {
    __shared__ __bf16 shA[16 * (MAXKDPAD + AROWPAD)];

    const int lane  = threadIdx.x & 31;
    const int wave  = threadIdx.x >> 5;
    const int tileM = blockIdx.y;
    const int rowStride = KdPad + AROWPAD;

    // ---- async-stage this block's 16-row weight tile into LDS (b32 chunks) --
    {
        const int dwPerRow = KdPad >> 1;           // bf16 pairs per row
        const int totalDw  = 16 * dwPerRow;
        for (int i = threadIdx.x; i < totalDw; i += 128) {
            int r   = i / dwPerRow;
            int col = i - r * dwPerRow;
            const __bf16* g = Wp + (size_t)(tileM * 16 + r) * KdPad + col * 2;
            unsigned dst = lds_addr_of(&shA[r * rowStride + col * 2]);
            asm volatile("global_load_async_to_lds_b32 %0, %1, off"
                         :: "v"(dst), "v"(g) : "memory");
        }
        asm volatile("s_wait_asynccnt 0x0" ::: "memory");
    }
    __syncthreads();

    const int tilesN = (N + 15) >> 4;
    const int tileN0 = (blockIdx.x * 4 + wave) * 2;
    if (tileN0 >= tilesN) return;                  // uniform per wave

    const int grp = lane >> 4;                     // half-wave selector
    const int lm  = lane & 15;
    const int Kd  = IC * K;

    const int n0 = tileN0 * 16 + lm;
    const int n1 = n0 + 16;
    const bool nv0 = (n0 < N);
    const bool nv1 = (n1 < N);
    const int bb0 = nv0 ? (n0 / Tout) : 0;
    const int tt0 = nv0 ? (n0 - bb0 * Tout) : 0;
    const int bb1 = nv1 ? (n1 / Tout) : 0;
    const int tt1 = nv1 ? (n1 - bb1 * Tout) : 0;

    v8f acc0, acc1;
#pragma unroll
    for (int i = 0; i < 8; ++i) { acc0[i] = 0.0f; acc1[i] = 0.0f; }

    const __bf16* arow = &shA[lm * rowStride];

    for (int kd0 = 0; kd0 < KdPad; kd0 += 32) {
        // ---- A fragment from LDS: lanes 0-15 K∈{0..15}, 16-31 K∈{16..31}
        v16bf a;
        const __bf16* ar = arow + kd0;
#pragma unroll
        for (int e = 0; e < 16; ++e) {
            int kk = grp * 16 + e;
            a[e] = ar[kk];
        }
        const int kdStart = kd0 + grp * 16;
        v16bf b0 = gather_b<K>(X, bb0, tt0, IC, stride, pad, Tin, kdStart, Kd, nv0);
        v16bf b1 = gather_b<K>(X, bb1, tt1, IC, stride, pad, Tin, kdStart, Kd, nv1);
        acc0 = __builtin_amdgcn_wmma_f32_16x16x32_bf16(false, a, false, b0,
                                                       (short)0, acc0, false, false);
        acc1 = __builtin_amdgcn_wmma_f32_16x16x32_bf16(false, a, false, b1,
                                                       (short)0, acc1, false, false);
    }

    if (nv0) {
#pragma unroll
        for (int r = 0; r < 8; ++r) {
            int mm = tileM * 16 + r + grp * 8;     // C/D layout: VGPR r -> M = r + 8*grp
            Y[(bb0 * OC + mm) * Tout + tt0] = acc0[r] + bias[mm];
        }
    }
    if (nv1) {
#pragma unroll
        for (int r = 0; r < 8; ++r) {
            int mm = tileM * 16 + r + grp * 8;
            Y[(bb1 * OC + mm) * Tout + tt1] = acc1[r] + bias[mm];
        }
    }
}

// ---------------------------------------------------------------------------
// Naive conv for OC==1 (final decoder stage).
// ---------------------------------------------------------------------------
__global__ void conv1d_oc1_kernel(const float* __restrict__ w, const float* __restrict__ bias,
                                  const float* __restrict__ x, float* __restrict__ y,
                                  int IC, int k, int pad, int Tin, int total) {
    int i = blockIdx.x * blockDim.x + threadIdx.x;
    if (i >= total) return;
    int b = i / Tin, t = i - b * Tin;
    float acc = bias[0];
    for (int c = 0; c < IC; ++c) {
        const float* xp = x + (b * IC + c) * Tin;
        const float* wp = w + c * k;
        for (int kk = 0; kk < k; ++kk) {
            int ts = t + kk - pad;
            if (ts >= 0 && ts < Tin) acc += wp[kk] * xp[ts];
        }
    }
    y[i] = acc;
}

// ---------------------------------------------------------------------------
// GroupNorm(num_groups=1): per-sample mean/var over C*T, then affine (+leaky).
// ---------------------------------------------------------------------------
__global__ void meanvar_kernel(const float* __restrict__ x, float* __restrict__ stats, int n) {
    __shared__ float ss[256], sq[256];
    int b = blockIdx.x;
    const float* p = x + (size_t)b * n;
    float s = 0.0f, q = 0.0f;
    for (int i = threadIdx.x; i < n; i += blockDim.x) { float v = p[i]; s += v; q += v * v; }
    ss[threadIdx.x] = s; sq[threadIdx.x] = q;
    __syncthreads();
    for (int off = 128; off > 0; off >>= 1) {
        if ((int)threadIdx.x < off) { ss[threadIdx.x] += ss[threadIdx.x + off]; sq[threadIdx.x] += sq[threadIdx.x + off]; }
        __syncthreads();
    }
    if (threadIdx.x == 0) {
        float mu  = ss[0] / (float)n;
        float var = sq[0] / (float)n - mu * mu;
        stats[2 * b]     = mu;
        stats[2 * b + 1] = rsqrtf(var + GN_EPS);
    }
}

__global__ void gn_apply_kernel(const float* __restrict__ x, float* __restrict__ y,
                                const float* __restrict__ stats,
                                const float* __restrict__ gg, const float* __restrict__ bb,
                                int C, int T, int total, int do_leaky) {
    int i = blockIdx.x * blockDim.x + threadIdx.x;
    if (i >= total) return;
    int b = i / (C * T);
    int c = (i / T) % C;
    float mu = stats[2 * b], rs = stats[2 * b + 1];
    float v = (x[i] - mu) * rs * gg[c] + bb[c];
    if (do_leaky) v = (v > 0.0f) ? v : LEAKY_SLOPE * v;
    y[i] = v;
}

__global__ void add_leaky_kernel(const float* __restrict__ x, const float* __restrict__ h,
                                 float* __restrict__ y, int total) {
    int i = blockIdx.x * blockDim.x + threadIdx.x;
    if (i >= total) return;
    float v = x[i] + h[i];
    y[i] = (v > 0.0f) ? v : LEAKY_SLOPE * v;
}

// ---------------------------------------------------------------------------
// nn.Upsample(mode='linear', align_corners=False)
// ---------------------------------------------------------------------------
__global__ void upsample_kernel(const float* __restrict__ x, float* __restrict__ y,
                                int L, int s, int total) {
    int i = blockIdx.x * blockDim.x + threadIdx.x;
    if (i >= total) return;
    int Lo = L * s;
    int bc = i / Lo, to = i - bc * Lo;
    float pos = ((float)to + 0.5f) / (float)s - 0.5f;
    pos = fminf(fmaxf(pos, 0.0f), (float)(L - 1));
    int i0 = (int)floorf(pos);
    int i1 = min(i0 + 1, L - 1);
    float w = pos - (float)i0;
    const float* p = x + (size_t)bc * L;
    y[i] = p[i0] * (1.0f - w) + p[i1] * w;
}

// ---------------------------------------------------------------------------
// RVQ stage: fused distance + argmin + quantize + residual update + loss accum.
// residual/qtotal: (NB, 64, Tz); one block (256 thr) per (b, t) row.
// ---------------------------------------------------------------------------
__global__ __launch_bounds__(256)
void rvq_stage_kernel(float* __restrict__ residual, float* __restrict__ qtotal,
                      const float* __restrict__ cb, float* __restrict__ codes_out,
                      float* __restrict__ loss_acc, int Tz) {
    __shared__ float frow[64];
    __shared__ float sval[256];
    __shared__ int   sidx[256];
    int row = blockIdx.x;
    int b = row / Tz, t = row - b * Tz;
    int j = threadIdx.x;
    if (j < 64) frow[j] = residual[((size_t)b * 64 + j) * Tz + t];
    __syncthreads();

    float d = 0.0f;
    const float* cj = cb + j * 64;
    for (int c = 0; c < 64; ++c) { float df = frow[c] - cj[c]; d += df * df; }
    sval[j] = d; sidx[j] = j;
    __syncthreads();
    for (int off = 128; off > 0; off >>= 1) {
        if (j < off) {
            if (sval[j + off] < sval[j] || (sval[j + off] == sval[j] && sidx[j + off] < sidx[j])) {
                sval[j] = sval[j + off]; sidx[j] = sidx[j + off];
            }
        }
        __syncthreads();
    }
    int best = sidx[0];
    __syncthreads();

    float part = 0.0f;
    if (j < 64) {
        float q = cb[best * 64 + j];
        float r = frow[j];
        float df = q - r;
        part = df * df;
        size_t off = ((size_t)b * 64 + j) * Tz + t;
        qtotal[off]  += q;
        residual[off] = r - q;
    }
    sval[j] = part;
    __syncthreads();
    for (int off = 128; off > 0; off >>= 1) {
        if (j < off) sval[j] += sval[j + off];
        __syncthreads();
    }
    if (j == 0) {
        atomicAdd(loss_acc, sval[0]);
        codes_out[b * (NCB * Tz) + t] = (float)best;   // codes as fp32 in d_out
    }
}

// ---------------------------------------------------------------------------
// pooled = concat(mean, std) over last axis of feats (NB, 64, LAT)
// ---------------------------------------------------------------------------
__global__ void pooled_kernel(const float* __restrict__ feats, float* __restrict__ pooled,
                              int C, int T) {
    int i = blockIdx.x * blockDim.x + threadIdx.x;
    if (i >= NB * C) return;
    int b = i / C, c = i - b * C;
    const float* p = feats + ((size_t)b * C + c) * T;
    float s = 0.0f, q = 0.0f;
    for (int t = 0; t < T; ++t) { float v = p[t]; s += v; q += v * v; }
    float mu = s / (float)T;
    float var = q / (float)T - mu * mu;
    pooled[b * 2 * C + c]     = mu;
    pooled[b * 2 * C + C + c] = sqrtf(fmaxf(var, 0.0f));
}

// ---------------------------------------------------------------------------
// Small dense layers (VALU; tiny shapes)
// ---------------------------------------------------------------------------
__global__ void linear_kernel(const float* __restrict__ x, const float* __restrict__ w,
                              const float* __restrict__ bias, float* __restrict__ y,
                              int R, int I, int O, int do_leaky) {
    int i = blockIdx.x * blockDim.x + threadIdx.x;
    if (i >= R * O) return;
    int r = i / O, o = i - r * O;
    float acc = bias[o];
    const float* xp = x + r * I;
    const float* wp = w + o * I;
    for (int jj = 0; jj < I; ++jj) acc += xp[jj] * wp[jj];
    if (do_leaky) acc = (acc > 0.0f) ? acc : LEAKY_SLOPE * acc;
    y[i] = acc;
}

__global__ void layernorm_leaky_kernel(const float* __restrict__ x, float* __restrict__ y,
                                       const float* __restrict__ gg, const float* __restrict__ bb,
                                       int D) {
    __shared__ float ss[128], sq[128];
    int r = blockIdx.x, t = threadIdx.x;
    float v = x[r * D + t];
    ss[t] = v; sq[t] = v * v;
    __syncthreads();
    for (int off = 64; off > 0; off >>= 1) {
        if (t < off) { ss[t] += ss[t + off]; sq[t] += sq[t + off]; }
        __syncthreads();
    }
    float mu = ss[0] / (float)D;
    float var = sq[0] / (float)D - mu * mu;
    float o = (v - mu) * rsqrtf(var + GN_EPS) * gg[t] + bb[t];
    y[r * D + t] = (o > 0.0f) ? o : LEAKY_SLOPE * o;
}

__global__ void zero_kernel(float* p, int n) {
    int i = blockIdx.x * blockDim.x + threadIdx.x;
    if (i < n) p[i] = 0.0f;
}

__global__ void finalize_loss_kernel(const float* __restrict__ acc,
                                     float* __restrict__ out_sum, float* __restrict__ out_stages) {
    if (threadIdx.x == 0 && blockIdx.x == 0) {
        const float scale = 1.25f / ((float)NB * 64.0f * (float)LAT);
        float s = 0.0f;
        for (int i = 0; i < 4; ++i) { float l = acc[i] * scale; out_stages[i] = l; s += l; }
        *out_sum = s;
    }
}

// ---------------------------------------------------------------------------
// Host-side orchestration helpers
// ---------------------------------------------------------------------------
static void conv_wmma(hipStream_t s, const float* w, const float* bias, const float* x, float* y,
                      int IC, int k, int pad, int stride, int Tin, int Tout, int OC, __bf16* wpack) {
    int Kd = IC * k;
    int KdPad = ceil_div_i(Kd, 32) * 32;
    int nPack = OC * KdPad;
    pack_w_bf16_kernel<<<ceil_div_i(nPack, 256), 256, 0, s>>>(w, wpack, Kd, KdPad, OC);
    int N = NB * Tout;
    int tilesN = ceil_div_i(N, 16);
    int pairs  = ceil_div_i(tilesN, 2);
    dim3 grid(ceil_div_i(pairs, 4), OC / 16);
    if (k == 7) {
        conv1d_wmma_kernel<7><<<grid, 128, 0, s>>>(wpack, bias, x, y, IC, pad, stride,
                                                   Tin, Tout, OC, KdPad, N);
    } else if (k == 9) {
        conv1d_wmma_kernel<9><<<grid, 128, 0, s>>>(wpack, bias, x, y, IC, pad, stride,
                                                   Tin, Tout, OC, KdPad, N);
    } else {
        conv1d_wmma_kernel<25><<<grid, 128, 0, s>>>(wpack, bias, x, y, IC, pad, stride,
                                                    Tin, Tout, OC, KdPad, N);
    }
}

static void group_norm_run(hipStream_t s, const float* x, float* y, const float* g, const float* b,
                           int C, int T, float* stats, int leaky) {
    meanvar_kernel<<<NB, 256, 0, s>>>(x, stats, C * T);
    int total = NB * C * T;
    gn_apply_kernel<<<ceil_div_i(total, 256), 256, 0, s>>>(x, y, stats, g, b, C, T, total, leaky);
}

// res_block: xin -> out (out may alias tmp1's buffer after tmp1 is consumed)
static void res_block_run(hipStream_t s, const float* xin, float* tmp1, float* tmp2, float* out,
                          int C, int T,
                          const float* c1b, const float* c1w, const float* c2b, const float* c2w,
                          const float* gn1b, const float* gn1g, const float* gn2b, const float* gn2g,
                          __bf16* wpack, float* stats) {
    conv_wmma(s, c1w, c1b, xin, tmp1, C, 7, 3, 1, T, T, C, wpack);
    group_norm_run(s, tmp1, tmp1, gn1g, gn1b, C, T, stats, 1);        // leaky(GN1) feeds conv2
    conv_wmma(s, c2w, c2b, tmp1, tmp2, C, 7, 3, 1, T, T, C, wpack);
    group_norm_run(s, tmp2, tmp2, gn2g, gn2b, C, T, stats, 0);
    int total = NB * C * T;
    add_leaky_kernel<<<ceil_div_i(total, 256), 256, 0, s>>>(xin, tmp2, out, total);
}

static void decode_run(hipStream_t s, const float* z, float* out, const float* const* P,
                       float* big, float* m0, float* m1, float* m2,
                       __bf16* wpack, float* stats) {
    static const int dst[5] = {2, 2, 4, 4, 5};
    static const int dic[5] = {64, 128, 128, 64, 32};
    static const int doc[5] = {128, 128, 64, 32, 1};
    const float* cur = z;
    int curL = LAT;
    for (int st = 0; st < 4; ++st) {
        int Lup = curL * dst[st];
        int utotal = NB * dic[st] * Lup;
        upsample_kernel<<<ceil_div_i(utotal, 256), 256, 0, s>>>(cur, big, curL, dst[st], utotal);
        const float* const* q = P + 11 + 12 * st;   // dec stage leaves
        conv_wmma(s, q[1], q[0], big, m0, dic[st], 9, 4, 1, Lup, Lup, doc[st], wpack);
        group_norm_run(s, m0, m1, q[3], q[2], doc[st], Lup, stats, 1);
        res_block_run(s, m1, m2, m0, m2, doc[st], Lup,
                      q[4], q[5], q[6], q[7], q[8], q[9], q[10], q[11], wpack, stats);
        cur = m2;
        curL = Lup;
    }
    // final: upsample x5 + conv (32 -> 1), no GN/res
    int Lup = curL * 5;                                  // 40000
    int utotal = NB * 32 * Lup;
    upsample_kernel<<<ceil_div_i(utotal, 256), 256, 0, s>>>(cur, big, curL, 5, utotal);
    int total = NB * Lup;
    conv1d_oc1_kernel<<<ceil_div_i(total, 256), 256, 0, s>>>(P[60], P[59], big, out,
                                                             32, 9, 4, Lup, total);
}

// ---------------------------------------------------------------------------
// kernel_launch
// ---------------------------------------------------------------------------
extern "C" void kernel_launch(void* const* d_in, const int* in_sizes, int n_in,
                              void* d_out, int out_size, void* d_ws, size_t ws_size,
                              hipStream_t stream) {
    if (n_in < 126) return;
    // jax pytree sorted-key flatten order:
    //   P[0]        codebooks (4,256,64)
    //   P[1..4]     ctrl_dec  l1.b l1.w l2.b l2.w
    //   P[5..10]    ctrl_enc  l1.b l1.w l2.b l2.w ln.b ln.g
    //   P[11..58]   dec stages 0..3: conv.b conv.w gn.b gn.g res.c1.b res.c1.w
    //                                res.c2.b res.c2.w res.gn1.b res.gn1.g res.gn2.b res.gn2.g
    //   P[59..60]   dec stage 4: conv.b conv.w
    //   P[61..120]  enc stages 0..4 (same 12-leaf pattern)
    //   P[121..124] metric l1.b l1.w l2.b l2.w
    //   P[125]      x (64,1,40000)
    const float* P[126];
    for (int i = 0; i < 126; ++i) P[i] = (const float*)d_in[i];
    const float* x = P[125];
    const float* codebooks = P[0];
    float* out = (float*)d_out;

    // ---- workspace carve-up (bytes) ----
    const size_t BIG_F  = (size_t)NB * 32 * 40000;   // 81,920,000
    const size_t MED_F  = (size_t)NB * 32 * 8000;    // 16,384,000
    const size_t LATF   = (size_t)NB * CDIM * LAT;   // 512,000
    size_t off = 0;
    auto alloc = [&](size_t bytes) -> void* {
        void* p = (char*)d_ws + off;
        off += bytes;
        off = (off + 255) & ~(size_t)255;
        return p;
    };
    float*  big    = (float*) alloc(BIG_F * 4);
    float*  m0     = (float*) alloc(MED_F * 4);
    float*  m1     = (float*) alloc(MED_F * 4);
    float*  m2     = (float*) alloc(MED_F * 4);
    float*  feats  = (float*) alloc(LATF * 4);
    float*  resid  = (float*) alloc(LATF * 4);
    float*  zseq   = (float*) alloc(LATF * 4);
    float*  coarse = (float*) alloc(LATF * 4);
    float*  z1     = (float*) alloc(NB * 128 * 4);
    float*  pooled = (float*) alloc(NB * 128 * 4);
    float*  stats  = (float*) alloc(2 * NB * 4);
    float*  lossac = (float*) alloc(16);
    __bf16* wpack  = (__bf16*)alloc(128 * MAXKDPAD * 2);
    if (off > ws_size) return;   // workspace too small; bail deterministically

    // ---- output offsets (fp32 elements) ----
    const size_t OFF_RSEQ  = 0;
    const size_t OFF_RCTRL = (size_t)NB * NT;            // 2,560,000
    const size_t OFF_ZCTRL = OFF_RCTRL + (size_t)NB * NT;
    const size_t OFF_CODES = OFF_ZCTRL + NB * 16;
    const size_t OFF_METR  = OFF_CODES + (size_t)NB * NCB * LAT;
    const size_t OFF_LSUM  = OFF_METR + NB * 8;
    const size_t OFF_LST   = OFF_LSUM + 1;

    // =============================== Encoder ===============================
    static const int eTin[5]  = {40000, 8000, 2000, 500, 250};
    static const int eTout[5] = {8000, 2000, 500, 250, 125};
    static const int eIC[5]   = {1, 32, 64, 128, 128};
    static const int eOC[5]   = {32, 64, 128, 128, 64};
    static const int eK[5]    = {25, 9, 9, 9, 9};
    static const int eS[5]    = {5, 4, 4, 2, 2};

    const float* cur = x;
    for (int st = 0; st < 5; ++st) {
        const float* const* q = P + 61 + 12 * st;
        conv_wmma(stream, q[1], q[0], cur, m0, eIC[st], eK[st], eK[st] / 2, eS[st],
                  eTin[st], eTout[st], eOC[st], wpack);
        group_norm_run(stream, m0, m1, q[3], q[2], eOC[st], eTout[st], stats, 1);
        res_block_run(stream, m1, m2, m0, m2, eOC[st], eTout[st],
                      q[4], q[5], q[6], q[7], q[8], q[9], q[10], q[11], wpack, stats);
        cur = m2;
    }
    // feats = cur (NB, 64, 125)
    hipMemcpyAsync(feats, cur, LATF * 4, hipMemcpyDeviceToDevice, stream);
    hipMemcpyAsync(resid, cur, LATF * 4, hipMemcpyDeviceToDevice, stream);

    // =============================== RVQ ===================================
    zero_kernel<<<ceil_div_i((int)LATF, 256), 256, 0, stream>>>(zseq, (int)LATF);
    zero_kernel<<<1, 32, 0, stream>>>(lossac, 4);
    for (int i = 0; i < NCB; ++i) {
        rvq_stage_kernel<<<NB * LAT, 256, 0, stream>>>(
            resid, zseq, codebooks + (size_t)i * CBSZ * CDIM,
            out + OFF_CODES + (size_t)i * LAT, lossac + i, LAT);
    }

    // ============================ Control heads ============================
    pooled_kernel<<<ceil_div_i(NB * 64, 256), 256, 0, stream>>>(feats, pooled, 64, LAT);
    // ctrl_enc: l2( leaky(ln( l1(pooled) )) )
    linear_kernel<<<ceil_div_i(NB * 128, 256), 256, 0, stream>>>(pooled, P[6], P[5], z1, NB, 128, 128, 0);
    layernorm_leaky_kernel<<<NB, 128, 0, stream>>>(z1, z1, P[10], P[9], 128);
    linear_kernel<<<ceil_div_i(NB * 16, 256), 256, 0, stream>>>(z1, P[8], P[7], out + OFF_ZCTRL, NB, 128, 16, 0);
    // ctrl_dec: coarse = l2( leaky(l1(z_ctrl)) )  -> (NB, 64, 125)
    linear_kernel<<<ceil_div_i(NB * 128, 256), 256, 0, stream>>>(out + OFF_ZCTRL, P[2], P[1], z1, NB, 16, 128, 1);
    linear_kernel<<<ceil_div_i(NB * 8000, 256), 256, 0, stream>>>(z1, P[4], P[3], coarse, NB, 128, 8000, 0);
    // metric head
    linear_kernel<<<ceil_div_i(NB * 128, 256), 256, 0, stream>>>(out + OFF_ZCTRL, P[122], P[121], pooled, NB, 16, 128, 1);
    linear_kernel<<<ceil_div_i(NB * 8, 256), 256, 0, stream>>>(pooled, P[124], P[123], out + OFF_METR, NB, 128, 8, 0);

    // =============================== Decoders ==============================
    decode_run(stream, zseq,   out + OFF_RSEQ,  P, big, m0, m1, m2, wpack, stats);
    decode_run(stream, coarse, out + OFF_RCTRL, P, big, m0, m1, m2, wpack, stats);

    // =============================== Losses ================================
    finalize_loss_kernel<<<1, 1, 0, stream>>>(lossac, out + OFF_LSUM, out + OFF_LST);

    (void)in_sizes; (void)out_size;
}
